// MultiHeadAttention_2448131358991
// MI455X (gfx1250) — compile-verified, matrix operand order checked by reading, not run on hardware
//
#include <hip/hip_runtime.h>
#include <hip/hip_bf16.h>
#include <cstdint>
#include <cstddef>

// ---------------------------------------------------------------------------
// MHA layer for MI455X (gfx1250): bf16 WMMA everywhere, flash attention core,
// double-buffered LDS staging, async global->LDS when the toolchain has it.
// ---------------------------------------------------------------------------

#define DEV __device__ __forceinline__

typedef __attribute__((ext_vector_type(16))) __bf16 v16bf;
typedef __attribute__((ext_vector_type(8)))  __bf16 v8bf;
typedef __attribute__((ext_vector_type(8)))  float  v8f;
typedef __attribute__((ext_vector_type(4)))  float  v4f;
typedef int v4i __attribute__((vector_size(16)));   // matches builtin param type

union FragBF { v16bf v; v8bf h[2]; };

static constexpr int D_MODEL = 1024;
static constexpr int N_HEADS = 16;
static constexpr int D_K     = 64;
static constexpr int BATCH   = 2;
static constexpr int SEQ     = 4096;
static constexpr int M_TOT   = BATCH * SEQ;   // 8192

#if defined(__has_builtin)
#if __has_builtin(__builtin_amdgcn_global_load_async_to_lds_b128)
#define USE_ASYNC_LDS 1
#endif
#endif
#ifndef USE_ASYNC_LDS
#define USE_ASYNC_LDS 0
#endif

#define AS_GLOBAL __attribute__((address_space(1)))
#define AS_SHARED __attribute__((address_space(3)))

// cast helpers: const T* (generic) -> v4i* in AS1 / AS3
#define GPTR(p) ((AS_GLOBAL v4i*)(AS_GLOBAL void*)(void*)(p))
#define LPTR(p) ((AS_SHARED v4i*)(AS_SHARED void*)(void*)(p))

// fp32 -> bf16, round to nearest even
DEV __bf16 f2bf(float f) {
  unsigned u = __float_as_uint(f);
  u += 0x7FFFu + ((u >> 16) & 1u);
  unsigned short s = (unsigned short)(u >> 16);
  __bf16 r;
  __builtin_memcpy(&r, &s, 2);
  return r;
}

// Load a 16x32 bf16 WMMA A/B fragment per the CDNA5 16-bit layout:
// lane l: row = row0 + (l&15); two 8-element K chunks at kBase + (l>>4)*8
// and kBase + 16 + (l>>4)*8  (each chunk = one 16B b128 load).
DEV v16bf ld_frag(const __bf16* base, int row0, int stride, int kBase) {
  const int lane = threadIdx.x & 31;
  const int lr = lane & 15, hi = lane >> 4;
  const __bf16* p = base + (size_t)(row0 + lr) * stride + kBase + hi * 8;
  FragBF f;
  f.h[0] = *(const v8bf*)(p);
  f.h[1] = *(const v8bf*)(p + 16);
  return f.v;
}

DEV v8f wmma_bf16(v16bf a, v16bf b, v8f c) {
  // (neg_a, A, neg_b, B, c_mod, C, reuse_a, reuse_b)
  return __builtin_amdgcn_wmma_f32_16x16x32_bf16(false, a, false, b, (short)0, c,
                                                 false, false);
}

// 16 contiguous elements -> two v8bf chunks (with fp32->bf16 conversion)
DEV void load16(const float* src, v8bf& c0, v8bf& c1) {
  v4f f0 = *(const v4f*)(src + 0);
  v4f f1 = *(const v4f*)(src + 4);
  v4f f2 = *(const v4f*)(src + 8);
  v4f f3 = *(const v4f*)(src + 12);
#pragma unroll
  for (int i = 0; i < 4; ++i) {
    c0[i] = f2bf(f0[i]); c0[i + 4] = f2bf(f1[i]);
    c1[i] = f2bf(f2[i]); c1[i + 4] = f2bf(f3[i]);
  }
}
DEV void load16(const __bf16* src, v8bf& c0, v8bf& c1) {
  c0 = *(const v8bf*)(src);
  c1 = *(const v8bf*)(src + 8);
}

// ---------------------------------------------------------------------------
// GEMM: C[m,n] = sum_k A[m,k] * W[n,k] + bias[n]      (einsum 'bsd,ed->bse')
// MODE 0: bf16 out, head-major     [b,h,s,d]   (Q, K)
// MODE 1: bf16 out, head-transpose [b,h,d,s]   (V -> feeds WMMA B directly)
// MODE 2: fp32 out, row-major      [m,n]       (final projection)
// Block: 128x128 tile, 8 waves, each wave 64x32 (4x2 WMMA tiles), K-step 32.
// Double-buffered LDS: one barrier/iter, next tile's global loads overlap WMMA.
// ---------------------------------------------------------------------------
template <int MODE, typename AT>
__global__ __launch_bounds__(256)
void gemm_bias_kernel(const AT* __restrict__ A, const float* __restrict__ W,
                      const float* __restrict__ bias, void* __restrict__ outp) {
  __shared__ __align__(16) __bf16 At[2][128 * 32];
  __shared__ __align__(16) __bf16 Wt[2][128 * 32];

  const int m0   = blockIdx.y * 128;
  const int n0   = blockIdx.x * 128;
  const int tid  = threadIdx.x;
  const int wave = tid >> 5;
  const int lane = tid & 31;
  const int lr = lane & 15, hi = lane >> 4;
  const int mw = (wave >> 2) * 64;  // 0 / 64
  const int nw = (wave & 3) * 32;   // 0 / 32 / 64 / 96

  v8f acc[4][2];
  const v8f vzero = {0.f, 0.f, 0.f, 0.f, 0.f, 0.f, 0.f, 0.f};
#pragma unroll
  for (int i = 0; i < 4; ++i)
#pragma unroll
    for (int j = 0; j < 2; ++j) acc[i][j] = vzero;

  const int row  = tid >> 1;         // 0..127
  const int colb = (tid & 1) * 16;   // 0 / 16 (K offset within tile)

  const AT*    aptr = A + (size_t)(m0 + row) * D_MODEL + colb;
  const float* wptr = W + (size_t)(n0 + row) * D_MODEL + colb;

  // prologue: stage first K-tile into registers
  v8bf a0, a1, w0, w1;
  load16(aptr, a0, a1);
  load16(wptr, w0, w1);

  int buf = 0;
  for (int k0 = 0; k0 < D_MODEL; k0 += 32) {
    __bf16* at = &At[buf][0];
    __bf16* wt = &Wt[buf][0];
    *(v8bf*)(&at[row * 32 + colb])     = a0;
    *(v8bf*)(&at[row * 32 + colb + 8]) = a1;
    *(v8bf*)(&wt[row * 32 + colb])     = w0;
    *(v8bf*)(&wt[row * 32 + colb + 8]) = w1;
    __syncthreads();

    // overlap: issue next tile's global loads while this tile's WMMAs run
    if (k0 + 32 < D_MODEL) {
      load16(aptr + (k0 + 32), a0, a1);
      load16(wptr + (k0 + 32), w0, w1);
      if (k0 + 64 < D_MODEL) {
        __builtin_prefetch(aptr + (k0 + 64), 0, 0);  // global_prefetch_b8
        __builtin_prefetch(wptr + (k0 + 64), 0, 0);
      }
    }

    v16bf bfr[2];
#pragma unroll
    for (int j = 0; j < 2; ++j) bfr[j] = ld_frag(wt, nw + j * 16, 32, 0);
#pragma unroll
    for (int i = 0; i < 4; ++i) {
      v16bf afr = ld_frag(at, mw + i * 16, 32, 0);
#pragma unroll
      for (int j = 0; j < 2; ++j) acc[i][j] = wmma_bf16(afr, bfr[j], acc[i][j]);
    }
    buf ^= 1;
  }

  // ---- epilogue: bias + scatter per output layout ----
#pragma unroll
  for (int j = 0; j < 2; ++j) {
    const int n  = n0 + nw + j * 16 + lr;
    const float bn = bias[n];
#pragma unroll
    for (int i = 0; i < 4; ++i) {
#pragma unroll
      for (int r = 0; r < 8; ++r) {
        const int m = m0 + mw + i * 16 + r + hi * 8;  // C layout: row r (+8 in hi half)
        float val = acc[i][j][r] + bn;
        if constexpr (MODE == 2) {
          ((float*)outp)[(size_t)m * D_MODEL + n] = val;
        } else {
          const int b = m >> 12, s = m & (SEQ - 1);
          const int h = n >> 6,  d = n & (D_K - 1);
          size_t idx;
          if constexpr (MODE == 0)
            idx = (((size_t)b * N_HEADS + h) * SEQ + s) * D_K + d;
          else
            idx = (((size_t)b * N_HEADS + h) * D_K + d) * SEQ + s;
          ((__bf16*)outp)[idx] = f2bf(val);
        }
      }
    }
  }
}

// ---------------------------------------------------------------------------
// Flash attention: per block 128 query rows of one (b,h); stream KV in 64-wide
// tiles, double-buffered. 8 waves x 16 query rows. Online softmax with
// width-16 xor reductions (wave32: rows 0-7 in lanes 0-15, 8-15 in 16-31).
// K/V tiles staged with GLOBAL_LOAD_ASYNC_TO_LDS_B128 when available.
// ---------------------------------------------------------------------------
__global__ __launch_bounds__(256)
void flash_attn_kernel(const __bf16* __restrict__ qh, const __bf16* __restrict__ kh,
                       const __bf16* __restrict__ vt, const int* __restrict__ maskp,
                       __bf16* __restrict__ o) {
  __shared__ __align__(16) __bf16 Kt[2][64 * 64];     // [kv][d]
  __shared__ __align__(16) __bf16 Vtl[2][64 * 64];    // [d][kv] (pre-transposed)
  __shared__ __align__(16) __bf16 Pb[8][16 * 64];     // per-wave P scratch
  __shared__ int maskT[2][64];

  const int bh   = blockIdx.y;
  const int b    = bh >> 4;
  const int q0   = blockIdx.x * 128;
  const int tid  = threadIdx.x;
  const int wave = tid >> 5;
  const int lane = tid & 31;
  const int lr = lane & 15, hi = lane >> 4;
  const int qr = q0 + wave * 16;

  // Q fragments stay in registers for the whole kernel (2 K-steps of 32 over d_k)
  const __bf16* qbase = qh + ((size_t)bh * SEQ + qr) * D_K;
  v16bf qf[2];
  qf[0] = ld_frag(qbase, 0, D_K, 0);
  qf[1] = ld_frag(qbase, 0, D_K, 32);

  const v8f vzero = {0.f, 0.f, 0.f, 0.f, 0.f, 0.f, 0.f, 0.f};
  v8f oacc[4];
#pragma unroll
  for (int i = 0; i < 4; ++i) oacc[i] = vzero;
  float rmax[8], rsum[8];
#pragma unroll
  for (int r = 0; r < 8; ++r) { rmax[r] = -3.0e38f; rsum[r] = 0.0f; }

  const int row  = tid >> 2;         // 0..63
  const int colb = (tid & 3) * 16;   // 0/16/32/48

  // per-thread staging pointers (advance by kv tile)
  const __bf16* kp = kh + ((size_t)bh * SEQ + row) * D_K + colb;
  const __bf16* vp = vt + ((size_t)bh * D_K + row) * SEQ + colb;
  const int*    mp = maskp + (size_t)b * SEQ;

  int buf = 0;

#if USE_ASYNC_LDS
  // ---- async staging: ASYNCcnt-tracked, no VGPR round trip ----
  auto stage = [&](int bsel, int kv0) {
    const __bf16* kg = kp + (size_t)kv0 * D_K;
    const __bf16* vg = vp + kv0;
    __bf16* kl = &Kt[bsel][row * 64 + colb];
    __bf16* vl = &Vtl[bsel][row * 64 + colb];
    __builtin_amdgcn_global_load_async_to_lds_b128(GPTR(kg),     LPTR(kl),     0, 0);
    __builtin_amdgcn_global_load_async_to_lds_b128(GPTR(kg + 8), LPTR(kl + 8), 0, 0);
    __builtin_amdgcn_global_load_async_to_lds_b128(GPTR(vg),     LPTR(vl),     0, 0);
    __builtin_amdgcn_global_load_async_to_lds_b128(GPTR(vg + 8), LPTR(vl + 8), 0, 0);
    if (tid < 64) maskT[bsel][tid] = mp[kv0 + tid];
  };
  stage(0, 0);
#else
  // ---- fallback: register-staged pipeline ----
  v8bf kr0, kr1, vr0, vr1;
  int mreg = 0;
  auto fetch = [&](int kv0) {
    kr0 = *(const v8bf*)(kp + (size_t)kv0 * D_K);
    kr1 = *(const v8bf*)(kp + (size_t)kv0 * D_K + 8);
    vr0 = *(const v8bf*)(vp + kv0);
    vr1 = *(const v8bf*)(vp + kv0 + 8);
    if (tid < 64) mreg = mp[kv0 + tid];
  };
  fetch(0);
#endif

  for (int kv0 = 0; kv0 < SEQ; kv0 += 64) {
#if USE_ASYNC_LDS
    asm volatile("s_wait_asynccnt 0x0" ::: "memory");
    __syncthreads();
    if (kv0 + 64 < SEQ) stage(buf ^ 1, kv0 + 64);
#else
    { // commit staged regs to this buffer
      __bf16* kl = &Kt[buf][row * 64 + colb];
      __bf16* vl = &Vtl[buf][row * 64 + colb];
      *(v8bf*)(kl)     = kr0;
      *(v8bf*)(kl + 8) = kr1;
      *(v8bf*)(vl)     = vr0;
      *(v8bf*)(vl + 8) = vr1;
      if (tid < 64) maskT[buf][tid] = mreg;
    }
    __syncthreads();
    if (kv0 + 64 < SEQ) fetch(kv0 + 64);   // overlaps WMMAs below
#endif

    const __bf16* kt  = &Kt[buf][0];
    const __bf16* vtl = &Vtl[buf][0];
    const int*    mk  = &maskT[buf][0];

    // ---- scores S = Q K^T * 1/sqrt(d_k), masked ----
    v8f st[4];
#pragma unroll
    for (int nt = 0; nt < 4; ++nt) {
      v8f s = vzero;
      s = wmma_bf16(qf[0], ld_frag(kt, nt * 16, 64, 0),  s);
      s = wmma_bf16(qf[1], ld_frag(kt, nt * 16, 64, 32), s);
      const bool keep = (mk[nt * 16 + lr] != 0);
#pragma unroll
      for (int r = 0; r < 8; ++r) s[r] = keep ? s[r] * 0.125f : -1.0e9f;
      st[nt] = s;
    }

    // ---- online softmax (per-row stats; width-16 xor reductions) ----
#pragma unroll
    for (int r = 0; r < 8; ++r) {
      float mx = fmaxf(fmaxf(st[0][r], st[1][r]), fmaxf(st[2][r], st[3][r]));
#pragma unroll
      for (int off = 1; off < 16; off <<= 1) mx = fmaxf(mx, __shfl_xor(mx, off, 16));
      const float newm = fmaxf(rmax[r], mx);
      const float corr = __expf(rmax[r] - newm);
      rmax[r] = newm;
      float ps = 0.0f;
#pragma unroll
      for (int nt = 0; nt < 4; ++nt) {
        const float p = __expf(st[nt][r] - newm);
        st[nt][r] = p;
        ps += p;
      }
#pragma unroll
      for (int off = 1; off < 16; off <<= 1) ps += __shfl_xor(ps, off, 16);
      rsum[r] = rsum[r] * corr + ps;
#pragma unroll
      for (int i = 0; i < 4; ++i) oacc[i][r] *= corr;
    }

    // ---- P -> wave-private LDS scratch (re-layout C-frag -> A-frag) ----
    __bf16* pb = &Pb[wave][0];
#pragma unroll
    for (int nt = 0; nt < 4; ++nt)
#pragma unroll
      for (int r = 0; r < 8; ++r)
        pb[(r + hi * 8) * 64 + nt * 16 + lr] = f2bf(st[nt][r]);
    // same-wave LDS ops are in-order: no barrier needed before reload

    // ---- O += P V ----
#pragma unroll
    for (int ks = 0; ks < 2; ++ks) {
      v16bf pf = ld_frag(pb, 0, 64, ks * 32);
#pragma unroll
      for (int dn = 0; dn < 4; ++dn) {
        v16bf vf = ld_frag(vtl, dn * 16, 64, ks * 32);
        oacc[dn] = wmma_bf16(pf, vf, oacc[dn]);
      }
    }
    buf ^= 1;
  }

  // ---- normalize and store O as bf16 [b, s, h*64+d] ----
#pragma unroll
  for (int dn = 0; dn < 4; ++dn) {
    const int dcol = (bh & 15) * D_K + dn * 16 + lr;
#pragma unroll
    for (int r = 0; r < 8; ++r) {
      const int srow = qr + r + hi * 8;
      o[((size_t)b * SEQ + srow) * D_MODEL + dcol] = f2bf(oacc[dn][r] / rsum[r]);
    }
  }
}

// ---------------------------------------------------------------------------
extern "C" void kernel_launch(void* const* d_in, const int* in_sizes, int n_in,
                              void* d_out, int out_size, void* d_ws, size_t ws_size,
                              hipStream_t stream) {
  (void)in_sizes; (void)n_in; (void)out_size; (void)ws_size;
  const float* q    = (const float*)d_in[0];
  const float* k    = (const float*)d_in[1];
  const float* v    = (const float*)d_in[2];
  const int*   mask = (const int*)d_in[3];
  const float* w_q  = (const float*)d_in[4];
  const float* b_q  = (const float*)d_in[5];
  const float* w_k  = (const float*)d_in[6];
  const float* b_k  = (const float*)d_in[7];
  const float* w_v  = (const float*)d_in[8];
  const float* b_v  = (const float*)d_in[9];
  const float* w_o  = (const float*)d_in[10];
  const float* b_o  = (const float*)d_in[11];

  const size_t headBytes = (size_t)BATCH * N_HEADS * SEQ * D_K * sizeof(unsigned short);
  char* ws = (char*)d_ws;
  __bf16* qhp = (__bf16*)(ws);
  __bf16* khp = (__bf16*)(ws + headBytes);
  __bf16* vtp = (__bf16*)(ws + 2 * headBytes);
  __bf16* op  = (__bf16*)(ws + 3 * headBytes);

  dim3 gg(D_MODEL / 128, M_TOT / 128);
  gemm_bias_kernel<0, float><<<gg, 256, 0, stream>>>(q, w_q, b_q, qhp);
  gemm_bias_kernel<0, float><<<gg, 256, 0, stream>>>(k, w_k, b_k, khp);
  gemm_bias_kernel<1, float><<<gg, 256, 0, stream>>>(v, w_v, b_v, vtp);

  dim3 fg(SEQ / 128, BATCH * N_HEADS);
  flash_attn_kernel<<<fg, 256, 0, stream>>>(qhp, khp, vtp, mask, op);

  gemm_bias_kernel<2, __bf16><<<gg, 256, 0, stream>>>(op, w_o, b_o, d_out);
}
